// QLayer_16578573762822
// MI455X (gfx1250) — 8-bit matrix operand order repaired and checked by reading, compiled, not run on hardware
//
#include <hip/hip_runtime.h>
#include <stdint.h>

typedef __attribute__((ext_vector_type(8))) int v8i;

#define N_ROWS 16384
#define K_DIM  4096
#define O_DIM  4096

#define KSTEP  128
#define BROW   144
#define BPANEL (128 * BROW)

union V8 {
  v8i  v;
  int  i[8];
  int2 d2[4];
  int4 d4[2];
};

__device__ __forceinline__ unsigned fenc(float f) {
  unsigned u = __float_as_uint(f);
  return (u & 0x80000000u) ? ~u : (u | 0x80000000u);
}
__device__ __forceinline__ float fdec(unsigned e) {
  unsigned u = (e & 0x80000000u) ? (e & 0x7FFFFFFFu) : ~e;
  return __uint_as_float(u);
}


__global__ void init_params(unsigned* prm) {
  if (threadIdx.x == 0) {
    prm[0] = 0xFFFFFFFFu; prm[1] = 0u;
    prm[2] = 0xFFFFFFFFu; prm[3] = 0u;
  }
}

__global__ __launch_bounds__(256) void minmax_x(const float4* __restrict__ x,
                                                unsigned* __restrict__ prm, int n4) {
  float mn = 0.f, mx = 0.f;
  for (int i = blockIdx.x * blockDim.x + threadIdx.x; i < n4;
       i += gridDim.x * blockDim.x) {
    float4 v = x[i];
    mn = fminf(mn, fminf(fminf(v.x, v.y), fminf(v.z, v.w)));
    mx = fmaxf(mx, fmaxf(fmaxf(v.x, v.y), fmaxf(v.z, v.w)));
  }
  for (int off = 16; off; off >>= 1) {
    mn = fminf(mn, __shfl_xor(mn, off, 32));
    mx = fmaxf(mx, __shfl_xor(mx, off, 32));
  }
  if ((threadIdx.x & 31) == 0) {
    atomicMin(&prm[0], fenc(mn));
    atomicMax(&prm[1], fenc(mx));
  }
}

__global__ void act_params(unsigned* prm) {
  if (threadIdx.x != 0) return;
  float mn = fminf(fdec(prm[0]), 0.f);
  float mx = fmaxf(fdec(prm[1]), 0.f);
  float sc = (mx - mn) / 255.f;
  float zp = rintf(-128.f - mn / sc);
  ((float*)prm)[4] = sc;
  ((float*)prm)[5] = zp;
  ((int*)prm)[6]   = (int)zp;
}

__global__ __launch_bounds__(256) void quant_act(const float4* __restrict__ x,
                                                 char4* __restrict__ qx,
                                                 const unsigned* __restrict__ prm, int n4) {
  const float sc = ((const float*)prm)[4];
  const float zp = ((const float*)prm)[5];
  for (int i = blockIdx.x * blockDim.x + threadIdx.x; i < n4;
       i += gridDim.x * blockDim.x) {
    float4 v = x[i];
    char4 q;
    q.x = (signed char)fminf(fmaxf(rintf(v.x / sc) + zp, -128.f), 127.f);
    q.y = (signed char)fminf(fmaxf(rintf(v.y / sc) + zp, -128.f), 127.f);
    q.z = (signed char)fminf(fmaxf(rintf(v.z / sc) + zp, -128.f), 127.f);
    q.w = (signed char)fminf(fmaxf(rintf(v.w / sc) + zp, -128.f), 127.f);
    qx[i] = q;
  }
}

__global__ __launch_bounds__(256) void quant_weight(const float* __restrict__ W,
                                                    float* __restrict__ wscale,
                                                    int* __restrict__ rowsum,
                                                    signed char* __restrict__ qw) {
  __shared__ float sred[8];
  __shared__ int   isum[8];
  __shared__ float s_ws;
  const int o = blockIdx.x;
  const float4* row = (const float4*)(W + (size_t)o * K_DIM);
  float m = 0.f;
  for (int i = threadIdx.x; i < K_DIM / 4; i += 256) {
    float4 v = row[i];
    m = fmaxf(m, fmaxf(fmaxf(fabsf(v.x), fabsf(v.y)), fmaxf(fabsf(v.z), fabsf(v.w))));
  }
  for (int off = 16; off; off >>= 1) m = fmaxf(m, __shfl_xor(m, off, 32));
  if ((threadIdx.x & 31) == 0) sred[threadIdx.x >> 5] = m;
  __syncthreads();
  if (threadIdx.x == 0) {
    float mm = sred[0];
    #pragma unroll
    for (int i = 1; i < 8; ++i) mm = fmaxf(mm, sred[i]);
    float wsv = mm / 127.f;
    s_ws = wsv;
    wscale[o] = wsv;
  }
  __syncthreads();
  const float wsv = s_ws;
  char4* qrow = (char4*)(qw + (size_t)o * K_DIM);
  int sum = 0;
  for (int i = threadIdx.x; i < K_DIM / 4; i += 256) {
    float4 v = row[i];
    char4 q;
    float a = fminf(fmaxf(rintf(v.x / wsv), -127.f), 127.f); q.x = (signed char)a; sum += (int)a;
    float b = fminf(fmaxf(rintf(v.y / wsv), -127.f), 127.f); q.y = (signed char)b; sum += (int)b;
    float c = fminf(fmaxf(rintf(v.z / wsv), -127.f), 127.f); q.z = (signed char)c; sum += (int)c;
    float d = fminf(fmaxf(rintf(v.w / wsv), -127.f), 127.f); q.w = (signed char)d; sum += (int)d;
    qrow[i] = q;
  }
  for (int off = 16; off; off >>= 1) sum += __shfl_xor(sum, off, 32);
  if ((threadIdx.x & 31) == 0) isum[threadIdx.x >> 5] = sum;
  __syncthreads();
  if (threadIdx.x == 0) {
    int s = 0;
    #pragma unroll
    for (int i = 0; i < 8; ++i) s += isum[i];
    rowsum[o] = s;
  }
}

__global__ __launch_bounds__(256) void corr_k(const float* __restrict__ bias,
                                              const float* __restrict__ wscale,
                                              const int* __restrict__ rowsum,
                                              int* __restrict__ corr,
                                              const unsigned* __restrict__ prm) {
  const int o = blockIdx.x * blockDim.x + threadIdx.x;
  if (o >= O_DIM) return;
  const float sc  = ((const float*)prm)[4];
  const int   zpi = ((const int*)prm)[6];
  corr[o] = (int)rintf(bias[o] / (wscale[o] * sc)) - zpi * rowsum[o];
}

__global__ __launch_bounds__(256) void gemm_q8_wmma(const signed char* __restrict__ qx,
                                                    const signed char* __restrict__ qw,
                                                    const int* __restrict__ corr,
                                                    float* __restrict__ res,
                                                    unsigned* __restrict__ prm) {
  __shared__ __align__(16) signed char btile[2][BPANEL];

  const int lane = threadIdx.x & 31;
  const int wid  = threadIdx.x >> 5;
  const int mblk = blockIdx.x >> 5;
  const int nblk = blockIdx.x & 31;
  const int mbase = (mblk << 7) + (wid << 4);
  const int obase = nblk << 7;
  const int half  = lane >> 4;
  const int l15   = lane & 15;

  unsigned gv[4], lb[4];
  #pragma unroll
  for (int j = 0; j < 4; ++j) {
    const int c  = threadIdx.x + (j << 8);
    const int ol = c >> 3;
    const int ko = (c & 7) << 4;
    gv[j] = (unsigned)((obase + ol) * K_DIM + ko);
    lb[j] = (unsigned)(ol * BROW + ko);
  }

#define ASYNC_B(BUF, KB)                                                      \
  do {                                                                        \
    _Pragma("unroll")                                                         \
    for (int j = 0; j < 4; ++j) {                                             \
      unsigned l_ = (unsigned)(uintptr_t)&btile[(BUF)][lb[j]];                \
      unsigned g_ = gv[j] + (unsigned)(KB);                                   \
      asm volatile("global_load_async_to_lds_b128 %0, %1, %2"                 \
                   :: "v"(l_), "v"(g_), "s"(qw) : "memory");                  \
    }                                                                         \
  } while (0)

#define LOAD_A(DST, OFF)                                                      \
  do {                                                                        \
    (DST).d2[0] = *(const int2*)(ap + (OFF));                                 \
    (DST).d2[1] = *(const int2*)(ap + (OFF) + 8);                             \
    (DST).d2[2] = *(const int2*)(ap + (OFF) + 32);                            \
    (DST).d2[3] = *(const int2*)(ap + (OFF) + 40);                            \
  } while (0)

  const signed char* ap = qx + (size_t)(mbase + l15) * K_DIM + (half << 4);

  const v8i zero = {0, 0, 0, 0, 0, 0, 0, 0};
  v8i acc[8];
  #pragma unroll
  for (int t = 0; t < 8; ++t) acc[t] = zero;

  ASYNC_B(0, 0);
  V8 a0, a1;
  LOAD_A(a0, 0);
  LOAD_A(a1, 64);
  asm volatile("s_wait_asynccnt 0" ::: "memory");
  __syncthreads();

  int cur = 0;
  for (int kb = 0; kb < K_DIM; kb += KSTEP) {
    const bool more = (kb + KSTEP < K_DIM);
    if (more) ASYNC_B(cur ^ 1, kb + KSTEP);
    V8 a0n, a1n;
    if (more) {
      LOAD_A(a0n, kb + KSTEP);
      LOAD_A(a1n, kb + KSTEP + 64);
    }

    const signed char* bt = &btile[cur][0];
    V8 bcur, bnxt;
    {
      const int rb0 = l15 * BROW + (half << 4);
      bcur.d4[0] = *(const int4*)&bt[rb0];
      bcur.d4[1] = *(const int4*)&bt[rb0 + 32];
    }
    #pragma unroll
    for (int u = 0; u < 16; ++u) {
      if (u + 1 < 16) {
        const int un = u + 1;
        const int rb = ((un & 7) * 16 + l15) * BROW + ((un >> 3) << 6) + (half << 4);
        bnxt.d4[0] = *(const int4*)&bt[rb];
        bnxt.d4[1] = *(const int4*)&bt[rb + 32];
      }
      const v8i av = (u < 8) ? a0.v : a1.v;
      acc[u & 7] = __builtin_amdgcn_wmma_i32_16x16x64_iu8(true, av, true, bcur.v,
                                                          acc[u & 7], false, false);
      if (u + 1 < 16) bcur = bnxt;
    }

    asm volatile("s_wait_asynccnt 0" ::: "memory");
    __syncthreads();
    cur ^= 1;
    if (more) { a0 = a0n; a1 = a1n; }
  }
#undef ASYNC_B
#undef LOAD_A

  unsigned emin = 0xFFFFFFFFu, emax = 0u;
  #pragma unroll
  for (int t = 0; t < 8; ++t) {
    V8 c; c.v = acc[t];
    const int col = obase + t * 16 + l15;
    const int cr  = corr[col];
    #pragma unroll
    for (int r = 0; r < 8; ++r) {
      const int row = mbase + (half << 3) + r;
      float v = (float)(c.i[r] + cr);
      res[(size_t)row * O_DIM + col] = v;
      unsigned e = fenc(v);
      emin = emin < e ? emin : e;
      emax = emax > e ? emax : e;
    }
  }

  for (int off = 16; off; off >>= 1) {
    unsigned om = __shfl_xor(emin, off, 32); emin = emin < om ? emin : om;
    unsigned ox = __shfl_xor(emax, off, 32); emax = emax > ox ? emax : ox;
  }
  if (lane == 0) {
    atomicMin(&prm[2], emin);
    atomicMax(&prm[3], emax);
  }
}

__global__ void requant_params(unsigned* prm) {
  if (threadIdx.x != 0) return;
  float rmn = fminf(fdec(prm[2]), 0.f);
  float rmx = fmaxf(fdec(prm[3]), 0.f);
  float re_sc = (rmx - rmn) / 255.f;
  float re_zp = rintf(-128.f - rmn / re_sc);
  ((float*)prm)[7] = re_sc;
  ((float*)prm)[8] = re_zp;
}

__global__ __launch_bounds__(256) void final_out(float4* __restrict__ res,
                                                 const float* __restrict__ wscale,
                                                 const unsigned* __restrict__ prm, int n4) {
  const float sc    = ((const float*)prm)[4];
  const float re_sc = ((const float*)prm)[7];
  const float re_zp = ((const float*)prm)[8];
  const float k     = sc * re_sc;
  for (int i = blockIdx.x * blockDim.x + threadIdx.x; i < n4;
       i += gridDim.x * blockDim.x) {
    float4 v = res[i];
    const int col = (i * 4) & (O_DIM - 1);
    float4 o;
    o.x = (fminf(fmaxf(rintf(v.x / re_sc) + re_zp, -128.f), 127.f) - re_zp) * (k * wscale[col + 0]);
    o.y = (fminf(fmaxf(rintf(v.y / re_sc) + re_zp, -128.f), 127.f) - re_zp) * (k * wscale[col + 1]);
    o.z = (fminf(fmaxf(rintf(v.z / re_sc) + re_zp, -128.f), 127.f) - re_zp) * (k * wscale[col + 2]);
    o.w = (fminf(fmaxf(rintf(v.w / re_sc) + re_zp, -128.f), 127.f) - re_zp) * (k * wscale[col + 3]);
    res[i] = o;
  }
}

extern "C" void kernel_launch(void* const* d_in, const int* in_sizes, int n_in,
                              void* d_out, int out_size, void* d_ws, size_t ws_size,
                              hipStream_t stream) {
  (void)in_sizes; (void)n_in; (void)out_size; (void)ws_size;
  const float* x    = (const float*)d_in[0];
  const float* W    = (const float*)d_in[1];
  const float* bias = (const float*)d_in[2];
  float* out = (float*)d_out;

  char* ws = (char*)d_ws;
  unsigned*    prm    = (unsigned*)ws;
  float*       wscale = (float*)(ws + 256);
  int*         rowsum = (int*)(ws + 16640);
  int*         corr   = (int*)(ws + 33024);
  signed char* qw     = (signed char*)(ws + 65536);
  signed char* qx     = (signed char*)(ws + 65536 + (16u << 20));

  const int n4 = (N_ROWS * K_DIM) / 4;

  init_params<<<1, 32, 0, stream>>>(prm);
  minmax_x<<<2048, 256, 0, stream>>>((const float4*)x, prm, n4);
  act_params<<<1, 32, 0, stream>>>(prm);
  quant_act<<<4096, 256, 0, stream>>>((const float4*)x, (char4*)qx, prm, n4);
  quant_weight<<<O_DIM, 256, 0, stream>>>(W, wscale, rowsum, qw);
  corr_k<<<O_DIM / 256, 256, 0, stream>>>(bias, wscale, rowsum, corr, prm);
  gemm_q8_wmma<<<(N_ROWS / 128) * (O_DIM / 128), 256, 0, stream>>>(qx, qw, corr, out, prm);
  requant_params<<<1, 32, 0, stream>>>(prm);
  final_out<<<4096, 256, 0, stream>>>((float4*)out, wscale, prm, n4);
}
